// LeFFPerformerLayer_42399917146396
// MI455X (gfx1250) — compile-verified, matrix operand order checked by reading, not run on hardware
//
#include <hip/hip_runtime.h>
#include <math.h>

// ---------------- types ----------------
typedef __bf16 bf16;
typedef __attribute__((ext_vector_type(16))) __bf16 v16bf;
typedef __attribute__((ext_vector_type(8)))  float  v8f;

// ---------------- problem constants ----------------
#define BATCH 8
#define TOK   4096            // N in reference
#define ROWS  (BATCH * TOK)   // 32768
#define DIN   512
#define EMBC  512
#define MRF   256             // random-feature dim m
#define HIDC  2048
#define RSQRT_M 0.0625f       // 1/sqrt(256)

__device__ __forceinline__ float gelu_exact(float x) {
    return 0.5f * x * (1.0f + erff(x * 0.70710678118654752f));
}

// ---------------- epilogue kinds ----------------
enum {
    EPI_STORE_BF16 = 0,   // kptv
    EPI_BIAS_BF16,        // k, q
    EPI_BIAS_F32_BF16,    // v (f32 for residual + bf16 for GEMM A)
    EPI_PRM,              // exp(acc - xd[row]) / sqrt(M) -> bf16
    EPI_DIVD,             // acc / (D[row]+eps) -> bf16
    EPI_BIAS_RES_F32,     // acc + bias + res -> f32 (attention out to d_out)
    EPI_BIAS_GELU_BF16,   // gelu(acc + bias) -> bf16
    EPI_BIAS_ADD_F32      // out_f32 += acc + bias (c3, in-place into d_out)
};

struct EpiParams {
    const float* bias;
    const float* xd;
    const float* Dv;
    const float* res;
    float*       outf;
    bf16*        outh;
    int          ldc;
};

struct Q2 { uint4 a, b; };

__device__ __forceinline__ v16bf load_frag(const bf16* rowbase) {
    Q2 q;
    q.a = *(const uint4*)(rowbase);
    q.b = *(const uint4*)(rowbase + 16);   // +16 bf16 elems = +32B (K 16..23 group)
    return __builtin_bit_cast(v16bf, q);
}

// Async copy of 32 bytes (2 x b128) from global to LDS, per lane.
// The 24-bit IOFFSET is added to BOTH the LDS and the global address
// (CDNA5 ISA pseudocode), so one address pair covers both halves.
__device__ __forceinline__ void async_copy_32B(const bf16* gsrc, bf16* lds_dst) {
    unsigned int loff = (unsigned int)(size_t)lds_dst;     // low 32b = group-segment offset
    unsigned long long ga = (unsigned long long)(size_t)gsrc;
    asm volatile("global_load_async_to_lds_b128 %0, %1, off"
                 :: "v"(loff), "v"(ga) : "memory");
    asm volatile("global_load_async_to_lds_b128 %0, %1, off offset:16"
                 :: "v"(loff), "v"(ga) : "memory");
}

__device__ __forceinline__ void wait_async_all() {
    asm volatile("s_wait_asynccnt 0x0" ::: "memory");
}

// ============================================================
//  Generic bf16 WMMA GEMM.
//  TRANS==false: C[M,N] = A[M,K] * B[N,K]^T   (both row-major)
//  TRANS==true : C[M,N] = A[K,M]^T * B[K,N]   (both row-major)
//  Block: 256 threads = 8 waves; tile 128(M) x 128(N); K step 32.
//  Wave (wm 0..3, wn 0..1) computes 32x64 via 2x4 of 16x16x32 WMMA.
// ============================================================
template<int EPI, bool TRANS>
__global__ __launch_bounds__(256)
void gemm_bf16(const bf16* __restrict__ A, const bf16* __restrict__ Bm,
               int Mdim, int Ndim, int Kdim, EpiParams p)
{
    __shared__ __align__(16) bf16 la[128][40];  // padded: 80B stride, 16B aligned
    __shared__ __align__(16) bf16 lb[128][40];

    const int tid  = threadIdx.x;
    const int bm   = blockIdx.y * 128;
    const int bn   = blockIdx.x * 128;
    const int lane = tid & 31;
    const int wave = tid >> 5;
    const int wm   = wave >> 1;     // 0..3  (32-row slice of M)
    const int wn   = wave & 1;      // 0..1  (64-col slice of N)
    const int half = lane >> 4;     // 0/1
    const int l15  = lane & 15;

    v8f acc[2][4] = {};

    for (int kb = 0; kb < Kdim; kb += 32) {
        __syncthreads();
        if (!TRANS) {
            // A tile 128x32 and B tile 128x32: each thread async-copies
            // 32B of each directly into LDS (no VGPR round-trip).
            int r = tid >> 1, c = (tid & 1) * 16;
            const bf16* gA = A + (size_t)(bm + r) * Kdim + kb + c;
            const bf16* gB = Bm + (size_t)(bn + r) * Kdim + kb + c;
            if (kb + 32 < Kdim) {
                __builtin_prefetch(gA + 32, 0, 1);
                __builtin_prefetch(gB + 32, 0, 1);
            }
            async_copy_32B(gA, &la[r][c]);
            async_copy_32B(gB, &lb[r][c]);
            wait_async_all();
        } else {
            // A is [K, M]: coalesced global read, transposed LDS scatter
            {
                int kr = tid >> 3, mc = (tid & 7) * 16;   // 32 k-rows x 8 chunks
                Q2 q;
                const uint4* src = (const uint4*)(A + (size_t)(kb + kr) * Mdim + bm + mc);
                q.a = src[0]; q.b = src[1];
                v16bf t = __builtin_bit_cast(v16bf, q);
                #pragma unroll
                for (int i = 0; i < 16; ++i) la[mc + i][kr] = t[i];
            }
            // B is [K, N]
            {
                int kr = tid >> 3, nc = (tid & 7) * 16;
                Q2 q;
                const uint4* src = (const uint4*)(Bm + (size_t)(kb + kr) * Ndim + bn + nc);
                q.a = src[0]; q.b = src[1];
                v16bf t = __builtin_bit_cast(v16bf, q);
                #pragma unroll
                for (int i = 0; i < 16; ++i) lb[nc + i][kr] = t[i];
            }
        }
        __syncthreads();

        v16bf fa[2], fb[4];
        #pragma unroll
        for (int mt = 0; mt < 2; ++mt)
            fa[mt] = load_frag(&la[wm * 32 + mt * 16 + l15][half * 8]);
        #pragma unroll
        for (int nt = 0; nt < 4; ++nt)
            fb[nt] = load_frag(&lb[wn * 64 + nt * 16 + l15][half * 8]);

        #pragma unroll
        for (int mt = 0; mt < 2; ++mt)
            #pragma unroll
            for (int nt = 0; nt < 4; ++nt)
                acc[mt][nt] = __builtin_amdgcn_wmma_f32_16x16x32_bf16(
                    false, fa[mt], false, fb[nt],
                    (short)0, acc[mt][nt], false, false);
    }

    // ---------------- epilogue ----------------
    #pragma unroll
    for (int mt = 0; mt < 2; ++mt) {
        #pragma unroll
        for (int nt = 0; nt < 4; ++nt) {
            const int gn = bn + wn * 64 + nt * 16 + l15;
            #pragma unroll
            for (int r = 0; r < 8; ++r) {
                const int gm = bm + wm * 32 + mt * 16 + half * 8 + r;
                float val = acc[mt][nt][r];
                const size_t idx = (size_t)gm * p.ldc + gn;
                if constexpr (EPI == EPI_STORE_BF16) {
                    p.outh[idx] = (bf16)val;
                } else if constexpr (EPI == EPI_BIAS_BF16) {
                    p.outh[idx] = (bf16)(val + p.bias[gn]);
                } else if constexpr (EPI == EPI_BIAS_F32_BF16) {
                    val += p.bias[gn];
                    p.outf[idx] = val;
                    p.outh[idx] = (bf16)val;
                } else if constexpr (EPI == EPI_PRM) {
                    p.outh[idx] = (bf16)(__expf(val - p.xd[gm]) * RSQRT_M);
                } else if constexpr (EPI == EPI_DIVD) {
                    p.outh[idx] = (bf16)(val / (p.Dv[gm] + 1e-8f));
                } else if constexpr (EPI == EPI_BIAS_RES_F32) {
                    p.outf[idx] = val + p.bias[gn] + p.res[idx];
                } else if constexpr (EPI == EPI_BIAS_GELU_BF16) {
                    p.outh[idx] = (bf16)gelu_exact(val + p.bias[gn]);
                } else if constexpr (EPI == EPI_BIAS_ADD_F32) {
                    p.outf[idx] += val + p.bias[gn];
                }
            }
        }
    }
}

// ============================================================
//  LayerNorm over width 512 (f32 in, bf16 out). One block per row.
// ============================================================
__global__ __launch_bounds__(256)
void ln_kernel(const float* __restrict__ x, const float* __restrict__ g,
               const float* __restrict__ bta, bf16* __restrict__ out)
{
    const int row = blockIdx.x;
    const int tid = threadIdx.x;
    const float* xr = x + (size_t)row * 512;
    float v0 = xr[tid], v1 = xr[tid + 256];

    __shared__ float sm[8], sm2[8];
    const int lane = tid & 31, w = tid >> 5;

    float s = v0 + v1;
    #pragma unroll
    for (int o = 16; o; o >>= 1) s += __shfl_down(s, o, 32);
    if (!lane) sm[w] = s;
    __syncthreads();
    float mu = 0.f;
    #pragma unroll
    for (int i = 0; i < 8; ++i) mu += sm[i];
    mu *= (1.0f / 512.0f);

    float d0 = v0 - mu, d1 = v1 - mu;
    float s2 = d0 * d0 + d1 * d1;
    #pragma unroll
    for (int o = 16; o; o >>= 1) s2 += __shfl_down(s2, o, 32);
    if (!lane) sm2[w] = s2;
    __syncthreads();
    float var = 0.f;
    #pragma unroll
    for (int i = 0; i < 8; ++i) var += sm2[i];
    var *= (1.0f / 512.0f);

    const float inv = rsqrtf(var + 1e-5f);
    out[(size_t)row * 512 + tid]       = (bf16)(d0 * inv * g[tid]       + bta[tid]);
    out[(size_t)row * 512 + tid + 256] = (bf16)(d1 * inv * g[tid + 256] + bta[tid + 256]);
}

// xd[row] = 0.5 * sum(x^2) over 512; one wave per row
__global__ __launch_bounds__(256)
void rowsumsq_kernel(const bf16* __restrict__ X, float* __restrict__ out)
{
    const int row  = blockIdx.x * 8 + (threadIdx.x >> 5);
    const int lane = threadIdx.x & 31;
    const bf16* xr = X + (size_t)row * 512;
    float s = 0.f;
    #pragma unroll 4
    for (int i = lane; i < 512; i += 32) { float v = (float)xr[i]; s += v * v; }
    #pragma unroll
    for (int o = 16; o; o >>= 1) s += __shfl_down(s, o, 32);
    if (!lane) out[row] = 0.5f * s;
}

__global__ __launch_bounds__(256)
void zero_kernel(float* __restrict__ p, int n)
{
    int i = blockIdx.x * 256 + threadIdx.x;
    if (i < n) p[i] = 0.f;
}

// partial ksum: grid (BATCH, 16); block sums 256 t-rows, atomically adds into ksum
__global__ __launch_bounds__(256)
void ksum_kernel(const bf16* __restrict__ kp, float* __restrict__ ks)
{
    const int b = blockIdx.x, chunk = blockIdx.y, m = threadIdx.x;
    const bf16* base = kp + ((size_t)b * TOK + (size_t)chunk * 256) * MRF + m;
    float s = 0.f;
    #pragma unroll 4
    for (int t = 0; t < 256; ++t) s += (float)base[(size_t)t * MRF];
    atomicAdd(&ks[b * MRF + m], s);
}

// D[row] = dot(qp[row,:], ksum[b,:]) over 256; one wave per row
__global__ __launch_bounds__(256)
void dvec_kernel(const bf16* __restrict__ qp, const float* __restrict__ ks,
                 float* __restrict__ Dv)
{
    const int row  = blockIdx.x * 8 + (threadIdx.x >> 5);
    const int lane = threadIdx.x & 31;
    const int b    = row >> 12;                  // 4096 rows per batch
    const bf16*  q = qp + (size_t)row * MRF;
    const float* k = ks + b * MRF;
    float s = 0.f;
    #pragma unroll
    for (int i = lane; i < MRF; i += 32) s += (float)q[i] * k[i];
    #pragma unroll
    for (int o = 16; o; o >>= 1) s += __shfl_down(s, o, 32);
    if (!lane) Dv[row] = s;
}

// depthwise 3x3 SAME conv on [B,64,64,2048] + bias + GELU; one block per pixel
__global__ __launch_bounds__(256)
void dwconv_kernel(const bf16* __restrict__ h1, const float* __restrict__ wgt,
                   const float* __restrict__ bias, bf16* __restrict__ h2)
{
    const int pix = blockIdx.x;            // b*4096 + h*64 + w
    const int b  = pix >> 12;
    const int hw = pix & 4095;
    const int hh = hw >> 6, ww = hw & 63;
    for (int c = threadIdx.x; c < HIDC; c += 256) {
        float acc = bias[c];
        #pragma unroll
        for (int ky = 0; ky < 3; ++ky) {
            const int ih = hh + ky - 1;
            if (ih < 0 || ih > 63) continue;
            #pragma unroll
            for (int kx = 0; kx < 3; ++kx) {
                const int iw = ww + kx - 1;
                if (iw < 0 || iw > 63) continue;
                acc += (float)h1[((((size_t)b * 64 + ih) * 64) + iw) * HIDC + c]
                       * wgt[(ky * 3 + kx) * HIDC + c];
            }
        }
        h2[(size_t)pix * HIDC + c] = (bf16)gelu_exact(acc);
    }
}

__global__ __launch_bounds__(256)
void cast_kernel(const float* __restrict__ in, bf16* __restrict__ out, int n)
{
    for (int i = blockIdx.x * 256 + threadIdx.x; i < n; i += gridDim.x * 256)
        out[i] = (bf16)in[i];
}

// ============================================================
//  Host orchestration
// ============================================================
extern "C" void kernel_launch(void* const* d_in, const int* in_sizes, int n_in,
                              void* d_out, int out_size, void* d_ws, size_t ws_size,
                              hipStream_t stream)
{
    (void)in_sizes; (void)n_in; (void)out_size; (void)ws_size;

    const float* x      = (const float*)d_in[0];
    const float* w      = (const float*)d_in[1];
    const float* kqv_w  = (const float*)d_in[2];
    const float* kqv_b  = (const float*)d_in[3];
    const float* proj_w = (const float*)d_in[4];
    const float* proj_b = (const float*)d_in[5];
    const float* ln1_g  = (const float*)d_in[6];
    const float* ln1_b  = (const float*)d_in[7];
    const float* ln2_g  = (const float*)d_in[8];
    const float* ln2_b  = (const float*)d_in[9];
    const float* c1_w   = (const float*)d_in[10];
    const float* c1_b   = (const float*)d_in[11];
    const float* c2_w   = (const float*)d_in[12];
    const float* c2_b   = (const float*)d_in[13];
    const float* c3_w   = (const float*)d_in[14];
    const float* c3_b   = (const float*)d_in[15];
    float* out = (float*)d_out;

    // -------- workspace carve-up (256B aligned) --------
    size_t off = 0;
    auto alloc = [&](size_t bytes) -> void* {
        void* p = (char*)d_ws + off;
        off = (off + bytes + 255) & ~(size_t)255;
        return p;
    };
    bf16*  xn      = (bf16*)alloc((size_t)ROWS * 512 * 2);
    bf16*  k_bf    = (bf16*)alloc((size_t)ROWS * 512 * 2);
    bf16*  q_bf    = (bf16*)alloc((size_t)ROWS * 512 * 2);
    float* v_f32   = (float*)alloc((size_t)ROWS * 512 * 4);
    bf16*  v_bf    = (bf16*)alloc((size_t)ROWS * 512 * 2);
    float* xdk     = (float*)alloc((size_t)ROWS * 4);
    float* xdq     = (float*)alloc((size_t)ROWS * 4);
    bf16*  kp_bf   = (bf16*)alloc((size_t)ROWS * MRF * 2);
    bf16*  qp_bf   = (bf16*)alloc((size_t)ROWS * MRF * 2);
    float* ksum    = (float*)alloc((size_t)BATCH * MRF * 4);
    float* Dv      = (float*)alloc((size_t)ROWS * 4);
    bf16*  kptv_bf = (bf16*)alloc((size_t)BATCH * 512 * MRF * 2);
    bf16*  y1_bf   = (bf16*)alloc((size_t)ROWS * 512 * 2);
    bf16*  yn      = (bf16*)alloc((size_t)ROWS * 512 * 2);
    bf16*  h1_bf   = (bf16*)alloc((size_t)ROWS * HIDC * 2);
    bf16*  h2_bf   = (bf16*)alloc((size_t)ROWS * HIDC * 2);
    bf16*  wq_bf   = (bf16*)alloc((size_t)1536 * 512 * 2);
    bf16*  w_bf    = (bf16*)alloc((size_t)MRF * 512 * 2);
    bf16*  pj_bf   = (bf16*)alloc((size_t)512 * 512 * 2);
    bf16*  c1_bf   = (bf16*)alloc((size_t)HIDC * 512 * 2);
    bf16*  c3_bf   = (bf16*)alloc((size_t)512 * HIDC * 2);

    const dim3 blk(256);

    // -------- weights -> bf16 --------
    cast_kernel<<<dim3(1024), blk, 0, stream>>>(kqv_w, wq_bf, 1536 * 512);
    cast_kernel<<<dim3(512),  blk, 0, stream>>>(w,     w_bf,  MRF * 512);
    cast_kernel<<<dim3(512),  blk, 0, stream>>>(proj_w, pj_bf, 512 * 512);
    cast_kernel<<<dim3(1024), blk, 0, stream>>>(c1_w,  c1_bf, HIDC * 512);
    cast_kernel<<<dim3(1024), blk, 0, stream>>>(c3_w,  c3_bf, 512 * HIDC);

    // -------- LN1 --------
    ln_kernel<<<dim3(ROWS), blk, 0, stream>>>(x, ln1_g, ln1_b, xn);

    // -------- kqv GEMMs (split k/q/v, weight rows [0:512), [512:1024), [1024:1536)) ----
    {
        EpiParams p{}; p.ldc = 512;
        p.bias = kqv_b; p.outh = k_bf;
        gemm_bf16<EPI_BIAS_BF16, false><<<dim3(4, 256), blk, 0, stream>>>(
            xn, wq_bf, ROWS, 512, 512, p);
        p.bias = kqv_b + 512; p.outh = q_bf;
        gemm_bf16<EPI_BIAS_BF16, false><<<dim3(4, 256), blk, 0, stream>>>(
            xn, wq_bf + (size_t)512 * 512, ROWS, 512, 512, p);
        p.bias = kqv_b + 1024; p.outh = v_bf; p.outf = v_f32;
        gemm_bf16<EPI_BIAS_F32_BF16, false><<<dim3(4, 256), blk, 0, stream>>>(
            xn, wq_bf + (size_t)1024 * 512, ROWS, 512, 512, p);
    }

    // -------- prm_exp: xd then kp/qp GEMMs with fused exp --------
    rowsumsq_kernel<<<dim3(ROWS / 8), blk, 0, stream>>>(k_bf, xdk);
    rowsumsq_kernel<<<dim3(ROWS / 8), blk, 0, stream>>>(q_bf, xdq);
    {
        EpiParams p{}; p.ldc = MRF;
        p.xd = xdk; p.outh = kp_bf;
        gemm_bf16<EPI_PRM, false><<<dim3(2, 256), blk, 0, stream>>>(
            k_bf, w_bf, ROWS, MRF, 512, p);
        p.xd = xdq; p.outh = qp_bf;
        gemm_bf16<EPI_PRM, false><<<dim3(2, 256), blk, 0, stream>>>(
            q_bf, w_bf, ROWS, MRF, 512, p);
    }

    // -------- D = qp . sum_t(kp) --------
    zero_kernel<<<dim3((BATCH * MRF + 255) / 256), blk, 0, stream>>>(ksum, BATCH * MRF);
    ksum_kernel<<<dim3(BATCH, 16), blk, 0, stream>>>(kp_bf, ksum);
    dvec_kernel<<<dim3(ROWS / 8), blk, 0, stream>>>(qp_bf, ksum, Dv);

    // -------- kptv[b] = v^T kp  (TN GEMM, per batch) --------
    for (int b = 0; b < BATCH; ++b) {
        EpiParams p{}; p.ldc = MRF;
        p.outh = kptv_bf + (size_t)b * 512 * MRF;
        gemm_bf16<EPI_STORE_BF16, true><<<dim3(2, 4), blk, 0, stream>>>(
            v_bf + (size_t)b * TOK * 512,
            kp_bf + (size_t)b * TOK * MRF,
            512, MRF, TOK, p);
    }

    // -------- y1[b] = qp @ kptv^T / (D+eps)  (NT GEMM, per batch) --------
    for (int b = 0; b < BATCH; ++b) {
        EpiParams p{}; p.ldc = 512;
        p.Dv = Dv + (size_t)b * TOK;
        p.outh = y1_bf + (size_t)b * TOK * 512;
        gemm_bf16<EPI_DIVD, false><<<dim3(4, 32), blk, 0, stream>>>(
            qp_bf + (size_t)b * TOK * MRF,
            kptv_bf + (size_t)b * 512 * MRF,
            TOK, 512, MRF, p);
    }

    // -------- attn out = v + y1 @ proj_w^T + proj_b  -> d_out (f32) --------
    {
        EpiParams p{}; p.ldc = 512;
        p.bias = proj_b; p.res = v_f32; p.outf = out;
        gemm_bf16<EPI_BIAS_RES_F32, false><<<dim3(4, 256), blk, 0, stream>>>(
            y1_bf, pj_bf, ROWS, 512, 512, p);
    }

    // -------- LN2 --------
    ln_kernel<<<dim3(ROWS), blk, 0, stream>>>(out, ln2_g, ln2_b, yn);

    // -------- LeFF c1 + GELU --------
    {
        EpiParams p{}; p.ldc = HIDC;
        p.bias = c1_b; p.outh = h1_bf;
        gemm_bf16<EPI_BIAS_GELU_BF16, false><<<dim3(16, 256), blk, 0, stream>>>(
            yn, c1_bf, ROWS, HIDC, 512, p);
    }

    // -------- depthwise 3x3 conv + bias + GELU --------
    dwconv_kernel<<<dim3(ROWS), blk, 0, stream>>>(h1_bf, c2_w, c2_b, h2_bf);

    // -------- LeFF c3 + bias, accumulate into d_out (residual) --------
    {
        EpiParams p{}; p.ldc = 512;
        p.bias = c3_b; p.outf = out;
        gemm_bf16<EPI_BIAS_ADD_F32, false><<<dim3(4, 256), blk, 0, stream>>>(
            h2_bf, c3_bf, ROWS, 512, HIDC, p);
    }
}
